// Seq_CNN_Big_Attention_89232240541893
// MI455X (gfx1250) — compile-verified, matrix operand order checked by reading, not run on hardware
//
#include <hip/hip_runtime.h>
#include <hip/hip_bf16.h>
#include <cstdint>
#include <cstddef>

// ---------------------------------------------------------------------------
// Shapes
// ---------------------------------------------------------------------------
#define NB 2
#define NN 4096
#define NM 8
#define NC 128
#define NH 8
#define ND 64
#define KQKV 640      // 5 taps * 128 in-channels
#define OCQKV 1536    // 512 (q) + 1024 (kv)
#define CC 4096       // H * 512 = attention-out row width
#define K2 20480      // 5 taps * 4096

typedef __attribute__((ext_vector_type(16))) __bf16 v16bf;
typedef __attribute__((ext_vector_type(8)))  float  v8f;
typedef __attribute__((ext_vector_type(4)))  unsigned int v4u;

union Frag16 {
  v16bf v;
  v4u   q[2];
};

__device__ __forceinline__ unsigned short f32_to_bf16(float f) {
  __bf16 h = (__bf16)f;
  return __builtin_bit_cast(unsigned short, h);
}

// One bf16 WMMA fragment = two contiguous 16B runs per lane:
// run0 at +((lane>>4)*8), run1 at +16 elements further (matches the
// 16-bit A/B VGPR striping in CDNA5 ISA).
__device__ __forceinline__ Frag16 load_frag(const unsigned short* p) {
  Frag16 f;
  f.q[0] = *(const v4u*)(p);
  f.q[1] = *(const v4u*)(p + 16);
  return f;
}

__device__ __forceinline__ v8f wmma_bf16(const Frag16& a, const Frag16& b, v8f c) {
  return __builtin_amdgcn_wmma_f32_16x16x32_bf16(false, a.v, false, b.v,
                                                 (short)0, c, false, false);
}

__device__ __forceinline__ int reflect_n(int g) {
  return g < 0 ? -g : (g > NN - 1 ? 2 * (NN - 1) - g : g);
}

// CDNA5 async global->LDS copy, 16B per lane, tracked by ASYNCcnt.
__device__ __forceinline__ void async_copy_b128(unsigned lds_off,
                                                const void* gptr) {
  unsigned long long ga = (unsigned long long)(uintptr_t)gptr;
  asm volatile("global_load_async_to_lds_b128 %0, %1, off"
               :: "v"(lds_off), "v"(ga)
               : "memory");
}
__device__ __forceinline__ void wait_asynccnt0() {
  asm volatile("s_wait_asynccnt 0" ::: "memory");
}

// ---------------------------------------------------------------------------
// Prep: f32 -> bf16 conversions into WMMA-friendly layouts
// ---------------------------------------------------------------------------
__global__ void prep_x_bf16(const float* __restrict__ x,
                            unsigned short* __restrict__ o, int n) {
  int i = blockIdx.x * 256 + threadIdx.x;
  if (i < n) o[i] = f32_to_bf16(x[i]);
}

// wqkvT[oc][t*128 + c], oc<512 from Wq, else from Wkv
__global__ void prep_wqkv(const float* __restrict__ Wq,
                          const float* __restrict__ Wkv,
                          unsigned short* __restrict__ o) {
  int i = blockIdx.x * 256 + threadIdx.x;
  if (i >= OCQKV * KQKV) return;
  int oc = i / KQKV, kk = i % KQKV;
  int t = kk >> 7, c = kk & 127;
  float v = (oc < 512) ? Wq[(size_t)oc * KQKV + c * 5 + t]
                       : Wkv[(size_t)(oc - 512) * KQKV + c * 5 + t];
  o[i] = f32_to_bf16(v);
}

// woutT[oc2][t*4096 + ic*8 + mm]
__global__ void prep_wout(const float* __restrict__ Wout,
                          unsigned short* __restrict__ o) {
  int i = blockIdx.x * 256 + threadIdx.x;
  if (i >= 1024 * K2) return;
  int oc2 = i / K2, kk = i % K2;
  int t = kk / CC, cc = kk % CC;
  int ic = cc >> 3, mm = cc & 7;
  o[i] = f32_to_bf16(Wout[((size_t)oc2 * 512 + ic) * 40 + mm * 5 + t]);
}

// ---------------------------------------------------------------------------
// Kernel 1: QKV projection conv-as-GEMM.  Per (b,m): [4096 x 640] x [640 x 1536]
// Block = 256 thr (8 waves); tile 32 rows x 256 out-channels; wave = 16x64.
// x tile (36 rows with reflect halo) staged in LDS via ASYNC global->LDS DMA.
// ---------------------------------------------------------------------------
__global__ __launch_bounds__(256) void qkv_conv_kernel(
    const unsigned short* __restrict__ xbf,
    const unsigned short* __restrict__ wqkvT,
    unsigned short* __restrict__ qmat,
    unsigned short* __restrict__ kmat,
    unsigned short* __restrict__ vtmat) {
  __shared__ unsigned short xtile[36 * 128];

  const int tid = threadIdx.x;
  const int itile = blockIdx.x * 32;
  const int octile = blockIdx.y * 256;
  const int bm = blockIdx.z;
  const int b = bm >> 3, m = bm & 7;

  // stage rows [itile-2, itile+33] (reflected) -> LDS, 16B async DMA per lane
  const unsigned xtile_base = (unsigned)(uintptr_t)(&xtile[0]);
  for (int e = tid; e < 36 * 16; e += 256) {
    int r = e >> 4, cv = e & 15;
    int g = reflect_n(itile - 2 + r);
    const unsigned short* src =
        xbf + ((size_t)(b * NN + g) * NM + m) * NC + cv * 8;
    async_copy_b128(xtile_base + (unsigned)(r * 256 + cv * 16), src);
  }
  wait_asynccnt0();
  __syncthreads();

  const int w = tid >> 5, lane = tid & 31;
  const int laneM = lane & 15, half8 = (lane >> 4) << 3;
  const int rh = w & 1, cg = w >> 1;
  const int rowbase = rh * 16;

  v8f acc[4] = {};
  for (int kc = 0; kc < 20; ++kc) {
    const int t = kc >> 2;
    const int cb = (kc & 3) * 32;
    Frag16 a = load_frag(&xtile[(rowbase + laneM + t) * 128 + cb + half8]);
#pragma unroll
    for (int blk = 0; blk < 4; ++blk) {
      const unsigned short* brow =
          wqkvT + (size_t)(octile + cg * 64 + blk * 16 + laneM) * KQKV +
          kc * 32 + half8;
      Frag16 bf = load_frag(brow);
      acc[blk] = wmma_bf16(a, bf, acc[blk]);
    }
  }

  // scatter to q (scaled), k, v-transposed; kk_attn = d*8 + m
#pragma unroll
  for (int blk = 0; blk < 4; ++blk) {
    const int oc = octile + cg * 64 + blk * 16 + laneM;
#pragma unroll
    for (int jv = 0; jv < 8; ++jv) {
      const int i = itile + rowbase + jv + half8;
      const float val = acc[blk][jv];
      if (oc < 512) {
        const int h = oc >> 6, d = oc & 63;
        qmat[((size_t)((b * NH + h) * NN) + i) * 512 + (d * 8 + m)] =
            f32_to_bf16(val * 0.125f);  // 1/sqrt(64) folded into q
      } else if (oc < 1024) {
        const int o = oc - 512, h = o >> 6, d = o & 63;
        kmat[((size_t)((b * NH + h) * NN) + i) * 512 + (d * 8 + m)] =
            f32_to_bf16(val);
      } else {
        const int o = oc - 1024, h = o >> 6, d = o & 63;
        vtmat[((size_t)(b * NH + h) * 512 + (d * 8 + m)) * NN + i] =
            f32_to_bf16(val);
      }
    }
  }
}

// ---------------------------------------------------------------------------
// Kernel 2: flash attention. Per (b,h): S = Q K^T (K=512), online softmax,
// O += P V.  Block = 256 thr (8 waves), 32-row i-tile, 64-col j-tiles.
// Phase A: each wave one 16x16 S block. Phase B: each wave a 16x128 O slice.
// ---------------------------------------------------------------------------
__global__ __launch_bounds__(256) void flash_attn_kernel(
    const unsigned short* __restrict__ qmat,
    const unsigned short* __restrict__ kmat,
    const unsigned short* __restrict__ vtmat,
    unsigned short* __restrict__ amat) {
  __shared__ float stile[32 * 64];
  __shared__ unsigned short ptile[32 * 64];
  __shared__ float rmax[32], rsum[32], ralpha[32], rinv[32];

  const int tid = threadIdx.x;
  const int itile = blockIdx.x * 32;
  const int bh = blockIdx.y;
  const int b = bh >> 3, h = bh & 7;

  const unsigned short* qb = qmat + (size_t)bh * NN * 512;
  const unsigned short* kb = kmat + (size_t)bh * NN * 512;
  const unsigned short* vb = vtmat + (size_t)bh * 512 * NN;

  const int w = tid >> 5, lane = tid & 31;
  const int laneM = lane & 15, half8 = (lane >> 4) << 3;
  const int rh = w & 1, colq = w >> 1;

  if (tid < 32) { rmax[tid] = -3.0e38f; rsum[tid] = 0.f; }
  v8f oacc[8] = {};
  __syncthreads();

  for (int jt = 0; jt < 64; ++jt) {
    const int j0 = jt * 64;
    // ---- Phase A: this wave's 16x16 block of S (rows rh*16, cols colq*16)
    {
      v8f s = {};
      const unsigned short* ar = qb + (size_t)(itile + rh * 16 + laneM) * 512;
      const unsigned short* br = kb + (size_t)(j0 + colq * 16 + laneM) * 512;
#pragma unroll
      for (int kc = 0; kc < 16; ++kc) {
        Frag16 a = load_frag(ar + kc * 32 + half8);
        Frag16 bf = load_frag(br + kc * 32 + half8);
        s = wmma_bf16(a, bf, s);
      }
#pragma unroll
      for (int jv = 0; jv < 8; ++jv)
        stile[(rh * 16 + jv + half8) * 64 + colq * 16 + laneM] = s[jv];
    }
    __syncthreads();
    // ---- Online softmax: wave 0, one lane per row
    if (tid < 32) {
      const int r = tid;
      float mold = rmax[r], mt = mold;
      for (int j = 0; j < 64; ++j) mt = fmaxf(mt, stile[r * 64 + j]);
      const float al = __expf(mold - mt);
      float sum = 0.f;
      for (int j = 0; j < 64; ++j) {
        const float p = __expf(stile[r * 64 + j] - mt);
        sum += p;
        ptile[r * 64 + j] = f32_to_bf16(p);
      }
      rsum[r] = rsum[r] * al + sum;
      rmax[r] = mt;
      ralpha[r] = al;
    }
    __syncthreads();
    // ---- Phase B: O = O*alpha + P V ; this wave: rows rh*16, cols colq*128
    {
      float a8[8];
#pragma unroll
      for (int jv = 0; jv < 8; ++jv) a8[jv] = ralpha[rh * 16 + jv + half8];
#pragma unroll
      for (int blk = 0; blk < 8; ++blk)
#pragma unroll
        for (int jv = 0; jv < 8; ++jv) oacc[blk][jv] *= a8[jv];
#pragma unroll
      for (int jc = 0; jc < 2; ++jc) {
        Frag16 a =
            load_frag(&ptile[(rh * 16 + laneM) * 64 + jc * 32 + half8]);
#pragma unroll
        for (int blk = 0; blk < 8; ++blk) {
          const int cc = colq * 128 + blk * 16 + laneM;
          Frag16 bf = load_frag(vb + (size_t)cc * NN + j0 + jc * 32 + half8);
          oacc[blk] = wmma_bf16(a, bf, oacc[blk]);
        }
      }
    }
    __syncthreads();
  }

  if (tid < 32) rinv[tid] = 1.0f / rsum[tid];
  __syncthreads();
  float inv8[8];
#pragma unroll
  for (int jv = 0; jv < 8; ++jv) inv8[jv] = rinv[rh * 16 + jv + half8];
#pragma unroll
  for (int blk = 0; blk < 8; ++blk) {
    const int ccfull = h * 512 + colq * 128 + blk * 16 + laneM;
#pragma unroll
    for (int jv = 0; jv < 8; ++jv) {
      const int i = itile + rh * 16 + jv + half8;
      amat[((size_t)b * NN + i) * CC + ccfull] =
          f32_to_bf16(oacc[blk][jv] * inv8[jv]);
    }
  }
}

// ---------------------------------------------------------------------------
// Kernel 3: output conv-as-GEMM.  Per b: [4096 x 20480] x [20480 x 1024].
// Output written in (b, oc2, i) flat order == reference's raw reshape.
// Weight stream prefetched ahead (global_prefetch) to hide L2 latency.
// ---------------------------------------------------------------------------
__global__ __launch_bounds__(256) void out_conv_kernel(
    const unsigned short* __restrict__ amat,
    const unsigned short* __restrict__ woutT,
    float* __restrict__ out) {
  const int tid = threadIdx.x;
  const int itile = blockIdx.x * 32;
  const int octile = blockIdx.y * 256;
  const int b = blockIdx.z;
  const int w = tid >> 5, lane = tid & 31;
  const int laneM = lane & 15, half8 = (lane >> 4) << 3;
  const int rh = w & 1, cg = w >> 1;

  const unsigned short* ab = amat + (size_t)b * NN * CC;
  v8f acc[4] = {};
  for (int kc = 0; kc < 640; ++kc) {
    const int t = kc >> 7;
    const int ccb = (kc & 127) * 32;
    const int gi = reflect_n(itile + rh * 16 + laneM + t - 2);
    if ((kc & 15) == 0 && kc < 624) {
      // prefetch this lane's weight row one 16-chunk group ahead
      __builtin_prefetch(
          woutT + (size_t)(octile + cg * 64 + laneM) * K2 + (kc + 16) * 32, 0,
          1);
    }
    Frag16 a = load_frag(ab + (size_t)gi * CC + ccb + half8);
#pragma unroll
    for (int blk = 0; blk < 4; ++blk) {
      const int oc2 = octile + cg * 64 + blk * 16 + laneM;
      Frag16 bf = load_frag(woutT + (size_t)oc2 * K2 + kc * 32 + half8);
      acc[blk] = wmma_bf16(a, bf, acc[blk]);
    }
  }
#pragma unroll
  for (int blk = 0; blk < 4; ++blk) {
    const int oc2 = octile + cg * 64 + blk * 16 + laneM;
#pragma unroll
    for (int jv = 0; jv < 8; ++jv) {
      const int i = itile + rh * 16 + jv + half8;
      out[((size_t)b * 1024 + oc2) * NN + i] = acc[blk][jv];
    }
  }
}

// ---------------------------------------------------------------------------
// Host launcher
// ---------------------------------------------------------------------------
extern "C" void kernel_launch(void* const* d_in, const int* in_sizes, int n_in,
                              void* d_out, int out_size, void* d_ws,
                              size_t ws_size, hipStream_t stream) {
  const float* x = (const float*)d_in[0];
  const float* Wq = (const float*)d_in[1];
  const float* Wkv = (const float*)d_in[2];
  const float* Wout = (const float*)d_in[3];
  float* out = (float*)d_out;

  char* ws = (char*)d_ws;
  size_t off = 0;
  auto alloc = [&](size_t bytes) -> void* {
    void* p = ws + off;
    off = (off + bytes + 255) & ~(size_t)255;
    return p;
  };
  const size_t xelems = (size_t)NB * NN * NM * NC;           // 8,388,608
  unsigned short* xbf   = (unsigned short*)alloc(xelems * 2);
  unsigned short* wqkvT = (unsigned short*)alloc((size_t)OCQKV * KQKV * 2);
  unsigned short* woutT = (unsigned short*)alloc((size_t)1024 * K2 * 2);
  unsigned short* qmat  = (unsigned short*)alloc((size_t)NB * NH * NN * 512 * 2);
  unsigned short* kmat  = (unsigned short*)alloc((size_t)NB * NH * NN * 512 * 2);
  unsigned short* vtmat = (unsigned short*)alloc((size_t)NB * NH * 512 * NN * 2);
  unsigned short* amat  = (unsigned short*)alloc((size_t)NB * NN * CC * 2);

  prep_x_bf16<<<dim3((unsigned)((xelems + 255) / 256)), dim3(256), 0, stream>>>(
      x, xbf, (int)xelems);
  prep_wqkv<<<dim3((OCQKV * KQKV + 255) / 256), dim3(256), 0, stream>>>(
      Wq, Wkv, wqkvT);
  prep_wout<<<dim3((1024 * K2 + 255) / 256), dim3(256), 0, stream>>>(Wout,
                                                                     woutT);

  qkv_conv_kernel<<<dim3(NN / 32, OCQKV / 256, NB * NM), dim3(256), 0,
                    stream>>>(xbf, wqkvT, qmat, kmat, vtmat);
  flash_attn_kernel<<<dim3(NN / 32, NB * NH), dim3(256), 0, stream>>>(
      qmat, kmat, vtmat, amat);
  out_conv_kernel<<<dim3(NN / 32, 1024 / 256, NB), dim3(256), 0, stream>>>(
      amat, woutT, out);
}
